// Encoder_44744969290566
// MI455X (gfx1250) — compile-verified, matrix operand order checked by reading, not run on hardware
//
#include <hip/hip_runtime.h>
#include <hip/hip_bf16.h>
#include <stdint.h>

// ---------------------------------------------------------------------------
// GIN encoder for MI455X (gfx1250). Pipeline:
//   buf0 = x + scatter_add(x)              (edge atomics)
//   buf1 = relu(buf0 @ W0^T + b0)          (WMMA bf16)
//   buf0 = relu(buf1 @ W1^T + b1)          (= z1)
//   buf1 = buf0 + scatter_add(buf0)
//   buf0 = relu(buf1 @ W2^T + b2)
//   out[0:NH]   = bn(relu(buf0 @ W3^T + b3))        (mode 2: ReLU->BN, = z)
//   out[NH:2NH] = prelu(bn_p(z @ Wp^T + bp))        (mode 1: BN->PReLU)
// ---------------------------------------------------------------------------

#define HDIM 128
#define LDSW 136  // padded u16 stride for W in LDS (272B rows -> bank shift 4)

typedef __bf16 v16bf __attribute__((ext_vector_type(16)));
typedef float  v8f   __attribute__((ext_vector_type(8)));

static __device__ __forceinline__ unsigned short f32_to_bf16(float f) {
  unsigned int u = __float_as_uint(f);
  unsigned int r = (u + 0x7FFFu + ((u >> 16) & 1u)) >> 16;  // RNE
  return (unsigned short)r;
}

// Hardware packed f32 -> 2xbf16 conversion (RNE). Builtin is not declared on
// this toolchain, so emit the gfx1250 instruction directly.
static __device__ __forceinline__ unsigned int cvt_pk_bf16(float lo, float hi) {
  unsigned int r;
  asm("v_cvt_pk_bf16_f32 %0, %1, %2" : "=v"(r) : "v"(lo), "v"(hi));
  return r;
}

union BF16Frag {
  v16bf v;
  unsigned short s[16];
  unsigned int u[8];
  uint4 q[2];
};

// --- small prep kernels -----------------------------------------------------

__global__ void prep_bn_params(const float* __restrict__ g, const float* __restrict__ b,
                               const float* __restrict__ m, const float* __restrict__ v,
                               const float* __restrict__ pg, const float* __restrict__ pb,
                               const float* __restrict__ pm, const float* __restrict__ pv,
                               float* __restrict__ out) {
  int i = threadIdx.x;  // 128 threads
  float s0 = g[i] * rsqrtf(v[i] + 1e-5f);
  out[i]       = s0;
  out[128 + i] = b[i] - m[i] * s0;
  float s1 = pg[i] * rsqrtf(pv[i] + 1e-5f);
  out[256 + i] = s1;
  out[384 + i] = pb[i] - pm[i] * s1;
}

__global__ void convert_weights(const float* __restrict__ w0, const float* __restrict__ w1,
                                const float* __restrict__ w2, const float* __restrict__ w3,
                                const float* __restrict__ w4, unsigned short* __restrict__ out) {
  int i = blockIdx.x * blockDim.x + threadIdx.x;
  if (i >= 5 * HDIM * HDIM) return;
  int which = i >> 14, pos = i & 16383;
  const float* src = which == 0 ? w0 : which == 1 ? w1 : which == 2 ? w2 : which == 3 ? w3 : w4;
  out[i] = f32_to_bf16(src[pos]);
}

__global__ void copy_f4(const float4* __restrict__ in, float4* __restrict__ out, int n4) {
  int i = blockIdx.x * blockDim.x + threadIdx.x;
  if (i < n4) out[i] = in[i];
}

// --- edge scatter: out[dst] += z[src], one thread per (edge, 4 features) ----

__global__ void scatter_add_edges(const float* __restrict__ z,
                                  const int* __restrict__ src_idx,
                                  const int* __restrict__ dst_idx,
                                  float* __restrict__ out, int nedges) {
  int idx = blockIdx.x * blockDim.x + threadIdx.x;
  int total = nedges * 32;
  if (idx >= total) return;
  int e = idx >> 5;
  int f = (idx & 31) << 2;
  int s = src_idx[e];
  int d = dst_idx[e];
  float4 v = *reinterpret_cast<const float4*>(z + (size_t)s * HDIM + f);
  float* o = out + (size_t)d * HDIM + f;
  unsafeAtomicAdd(o + 0, v.x);
  unsafeAtomicAdd(o + 1, v.y);
  unsafeAtomicAdd(o + 2, v.z);
  unsafeAtomicAdd(o + 3, v.w);
}

// --- fused GEMM: C = epilogue(A[nrows x128] @ W^T + bias) -------------------
// W held as bf16 row-major [out][in], staged whole in LDS.
// Block = 256 threads = 8 waves; wave w computes rows [blk*128 + w*16, +16) x 128 cols.
// Per wave: 4 K-chunks x 8 col-tiles = 32 x v_wmma_f32_16x16x32_bf16.
// mode 0: ReLU.
// mode 1: y = y*pscale + pshift; y = y>=0 ? y : a*y       (BN -> PReLU)
// mode 2: y = relu(y)*pscale + pshift                     (ReLU -> BN)

__global__ __launch_bounds__(256) void gemm_bias_epi(
    const float* __restrict__ A, const unsigned short* __restrict__ Wbf,
    const float* __restrict__ bias, float* __restrict__ C, int nrows, int mode,
    const float* __restrict__ pscale, const float* __restrict__ pshift,
    const float* __restrict__ prelu_a) {
  __shared__ unsigned short Wlds[HDIM * LDSW];

  const int tid = threadIdx.x;
  // Stage whole 128x128 bf16 W into LDS (padded stride to dodge bank conflicts).
  for (int i = tid; i < HDIM * HDIM; i += 256) {
    int r = i >> 7, c = i & 127;
    Wlds[r * LDSW + c] = Wbf[i];
  }
  __syncthreads();

  const int wave  = tid >> 5;
  const int lane  = tid & 31;
  const int row0  = blockIdx.x * 128 + wave * 16;
  const int mrow  = lane & 15;         // M (for A) / N (for B) within tile
  const int khalf = (lane >> 4) * 16;  // lanes 0-15: K base 0, lanes 16-31: K base 16

  v8f acc[8];
  const v8f vzero = {0.f, 0.f, 0.f, 0.f, 0.f, 0.f, 0.f, 0.f};
#pragma unroll
  for (int t = 0; t < 8; ++t) acc[t] = vzero;

  int arow = row0 + mrow;
  if (arow >= nrows) arow = nrows - 1;  // clamp loads; stores are guarded
  const float* Arow = A + (size_t)arow * HDIM;

#pragma unroll
  for (int kc = 0; kc < 4; ++kc) {
    const int k0 = kc * 32 + khalf;
    if (kc < 3) __builtin_prefetch(Arow + k0 + 32, 0, 3);  // global_prefetch_b8

    // A fragment: 16 contiguous f32 along K, packed-converted to bf16.
    BF16Frag af;
    const float4* ap = reinterpret_cast<const float4*>(Arow + k0);
#pragma unroll
    for (int i = 0; i < 4; ++i) {
      float4 f = ap[i];
      af.u[2 * i + 0] = cvt_pk_bf16(f.x, f.y);
      af.u[2 * i + 1] = cvt_pk_bf16(f.z, f.w);
    }

#pragma unroll
    for (int t = 0; t < 8; ++t) {
      // B fragment: B[k][n] = W[n][k]; lane holds W[t*16+mrow][k0 .. k0+15].
      BF16Frag bf;
      const uint4* bp =
          reinterpret_cast<const uint4*>(&Wlds[(t * 16 + mrow) * LDSW + k0]);
      bf.q[0] = bp[0];
      bf.q[1] = bp[1];
      acc[t] = __builtin_amdgcn_wmma_f32_16x16x32_bf16(
          false, af.v, false, bf.v, (short)0, acc[t], false, false);
    }
  }

  // Epilogue + store. C/D layout: VGPR r -> M = r (lanes 0-15) or 8+r (16-31), N = mrow.
  const float alpha = (mode == 1) ? prelu_a[0] : 0.f;
#pragma unroll
  for (int t = 0; t < 8; ++t) {
    const int col = t * 16 + mrow;
    const float bv = bias[col];
    const float sc = (mode != 0) ? pscale[col] : 1.f;
    const float sh = (mode != 0) ? pshift[col] : 0.f;
#pragma unroll
    for (int r = 0; r < 8; ++r) {
      const int row = row0 + ((lane < 16) ? r : 8 + r);
      if (row < nrows) {
        float v = acc[t][r] + bv;
        if (mode == 0) {
          v = v > 0.f ? v : 0.f;                 // ReLU
        } else if (mode == 2) {
          v = v > 0.f ? v : 0.f;                 // ReLU
          v = v * sc + sh;                       // BN (eval)
        } else {
          v = v * sc + sh;                       // BN (eval)
          v = v >= 0.f ? v : alpha * v;          // PReLU
        }
        C[(size_t)row * HDIM + col] = v;
      }
    }
  }
}

// ---------------------------------------------------------------------------

extern "C" void kernel_launch(void* const* d_in, const int* in_sizes, int n_in,
                              void* d_out, int out_size, void* d_ws, size_t ws_size,
                              hipStream_t stream) {
  const float* x      = (const float*)d_in[0];
  const int*   edge   = (const int*)d_in[1];
  const float* l0_w1  = (const float*)d_in[2];
  const float* l0_b1  = (const float*)d_in[3];
  const float* l0_w2  = (const float*)d_in[4];
  const float* l0_b2  = (const float*)d_in[5];
  const float* l1_w1  = (const float*)d_in[6];
  const float* l1_b1  = (const float*)d_in[7];
  const float* l1_w2  = (const float*)d_in[8];
  const float* l1_b2  = (const float*)d_in[9];
  const float* bn_g   = (const float*)d_in[10];
  const float* bn_b   = (const float*)d_in[11];
  const float* bn_m   = (const float*)d_in[12];
  const float* bn_v   = (const float*)d_in[13];
  const float* proj_w = (const float*)d_in[14];
  const float* proj_b = (const float*)d_in[15];
  const float* pbn_g  = (const float*)d_in[16];
  const float* pbn_b  = (const float*)d_in[17];
  const float* pbn_m  = (const float*)d_in[18];
  const float* pbn_v  = (const float*)d_in[19];
  const float* prelua = (const float*)d_in[20];

  const int N = in_sizes[0] / HDIM;
  const int E = in_sizes[1] / 2;
  const int NH = N * HDIM;

  // workspace layout
  char* ws = (char*)d_ws;
  float* bnp = (float*)ws;  // [bn_scale|bn_shift|pbn_scale|pbn_shift] x 128
  unsigned short* Wb = (unsigned short*)(ws + 4096);  // 5 x 128x128 bf16
  float* buf0 = (float*)(ws + 262144);
  float* buf1 = buf0 + (size_t)NH;

  unsigned short* W_l0w1 = Wb + 0 * 16384;
  unsigned short* W_l0w2 = Wb + 1 * 16384;
  unsigned short* W_l1w1 = Wb + 2 * 16384;
  unsigned short* W_l1w2 = Wb + 3 * 16384;
  unsigned short* W_proj = Wb + 4 * 16384;

  float* zout = (float*)d_out;        // bn(z2)
  float* pout = zout + (size_t)NH;    // projection head output

  const int n4 = NH / 4;
  const int cpyBlocks  = (n4 + 255) / 256;
  const int scatBlocks = (E * 32 + 255) / 256;
  const int gemmBlocks = (N + 127) / 128;
  const int wcvtBlocks = (5 * 16384 + 255) / 256;

  const int* e_src = edge;
  const int* e_dst = edge + E;

  // 0) parameter prep
  prep_bn_params<<<1, 128, 0, stream>>>(bn_g, bn_b, bn_m, bn_v,
                                        pbn_g, pbn_b, pbn_m, pbn_v, bnp);
  convert_weights<<<wcvtBlocks, 256, 0, stream>>>(l0_w1, l0_w2, l1_w1, l1_w2, proj_w, Wb);

  // 1) layer 0: h = x + agg(x)
  copy_f4<<<cpyBlocks, 256, 0, stream>>>((const float4*)x, (float4*)buf0, n4);
  scatter_add_edges<<<scatBlocks, 256, 0, stream>>>(x, e_src, e_dst, buf0, E);
  gemm_bias_epi<<<gemmBlocks, 256, 0, stream>>>(buf0, W_l0w1, l0_b1, buf1, N, 0,
                                                bnp, bnp + 128, prelua);
  gemm_bias_epi<<<gemmBlocks, 256, 0, stream>>>(buf1, W_l0w2, l0_b2, buf0, N, 0,
                                                bnp, bnp + 128, prelua);  // z1 = buf0

  // 2) layer 1: h = z1 + agg(z1)
  copy_f4<<<cpyBlocks, 256, 0, stream>>>((const float4*)buf0, (float4*)buf1, n4);
  scatter_add_edges<<<scatBlocks, 256, 0, stream>>>(buf0, e_src, e_dst, buf1, E);
  gemm_bias_epi<<<gemmBlocks, 256, 0, stream>>>(buf1, W_l1w1, l1_b1, buf0, N, 0,
                                                bnp, bnp + 128, prelua);
  // conv2 second linear + outer ReLU + encoder BN fused -> first output (z)
  gemm_bias_epi<<<gemmBlocks, 256, 0, stream>>>(buf0, W_l1w2, l1_b2, zout, N, 2,
                                                bnp, bnp + 128, prelua);

  // 3) projection head: Linear -> BN -> PReLU -> second output
  gemm_bias_epi<<<gemmBlocks, 256, 0, stream>>>(zout, W_proj, proj_b, pout, N, 1,
                                                bnp + 256, bnp + 384, prelua);
}
